// Attention_35880156790917
// MI455X (gfx1250) — matrix operand order repaired and checked by reading, not compiled, not run on hardware
//
#include <hip/hip_runtime.h>

typedef __bf16 bf16;
typedef __attribute__((ext_vector_type(16))) __bf16 v16bf;
typedef __attribute__((ext_vector_type(8)))  __bf16 v8bf;
typedef __attribute__((ext_vector_type(8)))  float  v8f;

#define HH 12
#define DHH 64
#define DDIM 768
#define SEQ 2048
#define BBATCH 4
#define MROWS (BBATCH*SEQ)            // 8192
#define MAX_LOG_SCALE 4.6051702f      // log(100)

// ---- helper: 16-lane-group sum (rows live across lanes 0-15 / 16-31) ----
__device__ __forceinline__ float rowred_sum(float x) {
#pragma unroll
  for (int m = 1; m < 16; m <<= 1) x += __shfl_xor(x, m, 32);
  return x;
}

union AV { v16bf v; v8bf h[2]; };

// ---------------- f32 -> bf16 convert ----------------
__global__ void cvt_f32_bf16(const float* __restrict__ in, bf16* __restrict__ out, int n) {
  int i = blockIdx.x * blockDim.x + threadIdx.x;
  if (i < n) out[i] = (bf16)in[i];
}

// ---------------- fused QKV projection + L2 norm (+scale) ----------------
// grid: (MROWS/128, HH, 3); block 256 (8 waves); wave computes 16 rows x 64 cols (one head)
__global__ __launch_bounds__(256)
void qkv_gemm(const bf16* __restrict__ X,
              const bf16* __restrict__ Wq, const bf16* __restrict__ Wk, const bf16* __restrict__ Wv,
              const float* __restrict__ bq, const float* __restrict__ bk, const float* __restrict__ bv,
              const float* __restrict__ logit_scale,
              bf16* __restrict__ qn, bf16* __restrict__ kn, bf16* __restrict__ vT)
{
  const int z    = blockIdx.z;                 // 0=Q 1=K 2=V
  const bf16* W  = (z == 0) ? Wq : (z == 1) ? Wk : Wv;
  const float* bias = (z == 0) ? bq : (z == 1) ? bk : bv;
  const int h    = blockIdx.y;
  const int n0   = h * DHH;
  const int wave = threadIdx.x >> 5;
  const int lane = threadIdx.x & 31;
  const int g    = lane >> 4;
  const int l    = lane & 15;
  const int m0   = blockIdx.x * 128 + wave * 16;

  v8f acc[4];
#pragma unroll
  for (int j = 0; j < 4; ++j)
#pragma unroll
    for (int r = 0; r < 8; ++r) acc[j][r] = 0.0f;

  const bf16* arow = X + (size_t)(m0 + l) * DDIM;
  for (int k0 = 0; k0 < DDIM; k0 += 32) {
    __builtin_prefetch(arow + k0 + 32, 0, 0);   // next A chunk -> global_prefetch_b8
    AV a;
    a.h[0] = *(const v8bf*)(arow + k0 + g * 16);       // K 0..7 / 16..23
    a.h[1] = *(const v8bf*)(arow + k0 + 8 + g * 16);   // K 8..15 / 24..31
    v16bf b[4];
#pragma unroll
    for (int j = 0; j < 4; ++j)
      b[j] = *(const v16bf*)(W + (size_t)(n0 + j * 16 + l) * DDIM + k0 + g * 16);
#pragma unroll
    for (int j = 0; j < 4; ++j)
      acc[j] = __builtin_amdgcn_wmma_f32_16x16x32_bf16(false, a.v, false, b[j],
                                                       (short)0, acc[j], false, false);
  }

  // bias (per-column, broadcast over rows)
#pragma unroll
  for (int j = 0; j < 4; ++j) {
    float bc = bias[n0 + j * 16 + l];
#pragma unroll
    for (int r = 0; r < 8; ++r) acc[j][r] += bc;
  }

  const int b_idx = m0 >> 11;         // row / 2048
  const int s_idx = m0 & 2047;

  if (z < 2) {
    // L2 normalize each 64-wide head row; fold logit scale into Q
    float sc = 1.0f;
    if (z == 0) sc = __expf(fminf(logit_scale[h], MAX_LOG_SCALE));
    bf16* dst = (z == 0) ? qn : kn;
#pragma unroll
    for (int r = 0; r < 8; ++r) {
      float ss = 0.0f;
#pragma unroll
      for (int j = 0; j < 4; ++j) ss += acc[j][r] * acc[j][r];
      ss = rowred_sum(ss);
      float inv = sc / fmaxf(sqrtf(ss), 1e-12f);
      int srow = s_idx + r + 8 * g;
      size_t base = (((size_t)b_idx * HH + h) * SEQ + srow) * DHH;
#pragma unroll
      for (int j = 0; j < 4; ++j)
        dst[base + j * 16 + l] = (bf16)(acc[j][r] * inv);
    }
  } else {
    // V stored transposed: vT[b][h][dh][s]; per lane 8 consecutive s -> one 16B store
    int s0 = s_idx + 8 * g;
#pragma unroll
    for (int j = 0; j < 4; ++j) {
      union { v8bf v; bf16 e[8]; } pk;
#pragma unroll
      for (int r = 0; r < 8; ++r) pk.e[r] = (bf16)acc[j][r];
      size_t off = (((size_t)b_idx * HH + h) * DHH + (j * 16 + l)) * SEQ + s0;
      *(v8bf*)(vT + off) = pk.v;
    }
  }
}

// ---------------- flash attention, cosine-bounded softmax ----------------
// scores = scale_h * cos(theta) <= scale_h exactly (scale folded into qn), so
// softmax can use the FIXED shift msub = scale_h: exp(s - msub) <= 1, no online
// max tracking, no accumulator rescale, and the denominator reduction moves to
// the epilogue (per-lane partials in-loop).
// grid: (SEQ/128, BBATCH*HH); block 256 (8 waves); wave owns a 16-row q tile
__global__ __launch_bounds__(256)
void attn(const bf16* __restrict__ qn, const bf16* __restrict__ kn,
          const bf16* __restrict__ vT, const float* __restrict__ logit_scale,
          bf16* __restrict__ ctx)
{
  __shared__ __align__(16) bf16 lds_p[8][16][32];  // per-wave P transpose staging

  const int wave = threadIdx.x >> 5;
  const int lane = threadIdx.x & 31;
  const int g    = lane >> 4;
  const int l    = lane & 15;
  const int bh   = blockIdx.y;                     // b*H + h
  const int s0   = blockIdx.x * 128 + wave * 16;
  const int h_idx = bh % HH;

  const float msub = __expf(fminf(logit_scale[h_idx], MAX_LOG_SCALE)); // == max score

  const size_t qkbase = (size_t)bh * SEQ * DHH;
  const bf16* kbase = kn + qkbase;
  const bf16* vbase = vT + (size_t)bh * DHH * SEQ;

  // q tile A-operands (loop invariant): dh 0..31 and 32..63
  AV qa0, qa1;
  {
    const bf16* qrow = qn + qkbase + (size_t)(s0 + l) * DHH;
    qa0.h[0] = *(const v8bf*)(qrow + 0  + g * 16);
    qa0.h[1] = *(const v8bf*)(qrow + 8  + g * 16);
    qa1.h[0] = *(const v8bf*)(qrow + 32 + g * 16);
    qa1.h[1] = *(const v8bf*)(qrow + 40 + g * 16);
  }

  v8f acc[4];
#pragma unroll
  for (int j = 0; j < 4; ++j)
#pragma unroll
    for (int r = 0; r < 8; ++r) acc[j][r] = 0.0f;
  float plsum[8];                                  // per-lane denominator partials
#pragma unroll
  for (int r = 0; r < 8; ++r) plsum[r] = 0.0f;

  v8f zc;
#pragma unroll
  for (int r = 0; r < 8; ++r) zc[r] = 0.0f;

  for (int kb = 0; kb < SEQ; kb += 32) {
    // prefetch next tiles into cache (global_prefetch_b8)
    {
      int kp = (kb + 32 < SEQ) ? kb + 32 : kb;
      __builtin_prefetch(kbase + (size_t)(kp + l) * DHH + g * 16, 0, 0);
      __builtin_prefetch(kbase + (size_t)(kp + 16 + l) * DHH + g * 16, 0, 0);
      __builtin_prefetch(vbase + (size_t)l * SEQ + kp + g * 16, 0, 0);
    }

    // ---- scores: S1 = q . k[kb..kb+15], S2 = q . k[kb+16..kb+31]  (K=64 as 2x32)
    v16bf kB;
    kB = *(const v16bf*)(kbase + (size_t)(kb + l) * DHH + 0 + g * 16);
    v8f S1 = __builtin_amdgcn_wmma_f32_16x16x32_bf16(false, qa0.v, false, kB, (short)0, zc, false, false);
    kB = *(const v16bf*)(kbase + (size_t)(kb + l) * DHH + 32 + g * 16);
    S1 = __builtin_amdgcn_wmma_f32_16x16x32_bf16(false, qa1.v, false, kB, (short)0, S1, false, false);
    kB = *(const v16bf*)(kbase + (size_t)(kb + 16 + l) * DHH + 0 + g * 16);
    v8f S2 = __builtin_amdgcn_wmma_f32_16x16x32_bf16(false, qa0.v, false, kB, (short)0, zc, false, false);
    kB = *(const v16bf*)(kbase + (size_t)(kb + 16 + l) * DHH + 32 + g * 16);
    S2 = __builtin_amdgcn_wmma_f32_16x16x32_bf16(false, qa1.v, false, kB, (short)0, S2, false, false);

    // ---- V B-operands (lane = dh column, contiguous keys from vT)
    v16bf vB[4];
#pragma unroll
    for (int j = 0; j < 4; ++j)
      vB[j] = *(const v16bf*)(vbase + (size_t)(j * 16 + l) * SEQ + kb + g * 16);

    // ---- fixed-shift softmax numerators + per-lane denominator partials
#pragma unroll
    for (int r = 0; r < 8; ++r) {
      float p1 = __expf(S1[r] - msub);
      float p2 = __expf(S2[r] - msub);
      S1[r] = p1; S2[r] = p2;
      plsum[r] += p1 + p2;
    }

    // ---- transpose P (D layout -> A layout) through this wave's LDS slice
#pragma unroll
    for (int r = 0; r < 8; ++r) {
      lds_p[wave][r + 8 * g][l]      = (bf16)S1[r];
      lds_p[wave][r + 8 * g][l + 16] = (bf16)S2[r];
    }
    asm volatile("" ::: "memory");   // keep DS write->read order (in-order within wave)
    AV pa;
    pa.h[0] = *(const v8bf*)&lds_p[wave][l][g * 16];
    pa.h[1] = *(const v8bf*)&lds_p[wave][l][8 + g * 16];
    asm volatile("" ::: "memory");

    // ---- accumulate ctx += P @ V
#pragma unroll
    for (int j = 0; j < 4; ++j)
      acc[j] = __builtin_amdgcn_wmma_f32_16x16x32_bf16(false, pa.v, false, vB[j],
                                                       (short)0, acc[j], false, false);
  }

  // ---- epilogue: one cross-lane reduction for the denominator, then store
  const int b_idx = bh / HH;
#pragma unroll
  for (int r = 0; r < 8; ++r) {
    float lsum = rowred_sum(plsum[r]);
    float inv = 1.0f / fmaxf(lsum, 1e-37f);
    int s = s0 + r + 8 * g;
    size_t rowoff = ((size_t)b_idx * SEQ + s) * DDIM + h_idx * DHH;
#pragma unroll
    for (int j = 0; j < 4; ++j)
      ctx[rowoff + j * 16 + l] = (bf16)(acc[j][r] * inv);
  }
}

// ---------------- output projection: out = ctx @ Wo^T + bo (f32 out) ----------------
// grid: (MROWS/128, HH); block 256
__global__ __launch_bounds__(256)
void out_gemm(const bf16* __restrict__ A, const bf16* __restrict__ W,
              const float* __restrict__ bias, float* __restrict__ out)
{
  const int wave = threadIdx.x >> 5;
  const int lane = threadIdx.x & 31;
  const int g    = lane >> 4;
  const int l    = lane & 15;
  const int n0   = blockIdx.y * 64;
  const int m0   = blockIdx.x * 128 + wave * 16;

  v8f acc[4];
#pragma unroll
  for (int j = 0; j < 4; ++j)
#pragma unroll
    for (int r = 0; r < 8; ++r) acc[j][r] = 0.0f;

  const bf16* arow = A + (size_t)(m0 + l) * DDIM;
  for (int k0 = 0; k0 < DDIM; k0 += 32) {
    __builtin_prefetch(arow + k0 + 32, 0, 0);
    AV a;
    a.h[0] = *(const v8bf*)(arow + k0 + g * 16);
    a.h[1] = *(const v8bf*)(arow + k0 + 8 + g * 16);
    v16bf b[4];
#pragma unroll
    for (int j = 0; j < 4; ++j)
      b[j] = *(const v16bf*)(W + (size_t)(n0 + j * 16 + l) * DDIM + k0 + g * 16);
#pragma unroll
    for (int j = 0; j < 4; ++j)
      acc[j] = __builtin_amdgcn_wmma_f32_16x16x32_bf16(false, a.v, false, b[j],
                                                       (short)0, acc[j], false, false);
  }

#pragma unroll
  for (int j = 0; j < 4; ++j) {
    float bc = bias[n0 + j * 16 + l];
#pragma unroll
    for (int r = 0; r < 8; ++r) {
      size_t off = (size_t)(m0 + r + 8 * g) * DDIM + n0 + j * 16 + l;
      out[off] = acc[j][r] + bc;
    }
  }
}

// ---------------- host launcher ----------------
extern "C" void kernel_launch(void* const* d_in, const int* in_sizes, int n_in,
                              void* d_out, int out_size, void* d_ws, size_t ws_size,
                              hipStream_t stream)
{
  const float* hs = (const float*)d_in[0];
  const float* Wq = (const float*)d_in[1];
  const float* bq = (const float*)d_in[2];
  const float* Wk = (const float*)d_in[3];
  const float* bk = (const float*)d_in[4];
  const float* Wv = (const float*)d_in[5];
  const float* bv = (const float*)d_in[6];
  const float* Wo = (const float*)d_in[7];
  const float* bo = (const float*)d_in[8];
  const float* ls = (const float*)d_in[9];
  float* out = (float*)d_out;

  // workspace carve-up (all bf16): X | Wq | Wk | Wv | Wo | qn | kn | vT | ctx
  bf16* ws  = (bf16*)d_ws;
  const size_t nX = (size_t)MROWS * DDIM;   // 6,291,456
  const size_t nW = (size_t)DDIM * DDIM;    //   589,824
  bf16* Xbf = ws;
  bf16* Wqb = Xbf + nX;
  bf16* Wkb = Wqb + nW;
  bf16* Wvb = Wkb + nW;
  bf16* Wob = Wvb + nW;
  bf16* qnb = Wob + nW;
  bf16* knb = qnb + nX;
  bf16* vTb = knb + nX;
  bf16* ctx = vTb + nX;

  cvt_f32_bf16<<<(int)((nX + 255) / 256), 256, 0, stream>>>(hs, Xbf, (int)nX);
  cvt_f32_bf16<<<(int)((nW + 255) / 256), 256, 0, stream>>>(Wq, Wqb, (int)nW);
  cvt_f32_bf16<<<(int)((nW + 255) / 256), 256, 0, stream>>>(Wk, Wkb, (int)nW);
  cvt_f32_bf16<<<(int)((nW + 255) / 256), 256, 0, stream>>>(Wv, Wvb, (int)nW);
  cvt_f32_bf16<<<(int)((nW + 255) / 256), 256, 0, stream>>>(Wo, Wob, (int)nW);

  qkv_gemm<<<dim3(MROWS / 128, HH, 3), 256, 0, stream>>>(
      Xbf, Wqb, Wkb, Wvb, bq, bk, bv, ls, qnb, knb, vTb);

  attn<<<dim3(SEQ / 128, BBATCH * HH), 256, 0, stream>>>(qnb, knb, vTb, ls, ctx);

  out_gemm<<<dim3(MROWS / 128, HH), 256, 0, stream>>>(ctx, Wob, bo, out);
}
